// ParentHomogeneousGNN_85255100825907
// MI455X (gfx1250) — matrix operand order repaired and checked by reading, not compiled, not run on hardware
//
#include <hip/hip_runtime.h>

#define DIM     256
#define NGRAPH  64
#define NPB     128      // nodes per block in pooling
#define WT_STR  264      // padded LDS row stride (elements): 16B-aligned rows, ~2-way banks
#define XS_STR  264

typedef __attribute__((ext_vector_type(16))) __bf16        v16bf;
typedef __attribute__((ext_vector_type(8)))  float         v8f;
typedef __attribute__((ext_vector_type(8)))  unsigned short ush8;

struct BF16Frag { ush8 lo, hi; };   // 32 bytes == v16bf

__device__ __forceinline__ unsigned short f32_to_bf16(float f) {
  unsigned int u = __builtin_bit_cast(unsigned int, f);
  u += 0x7FFFu + ((u >> 16) & 1u);          // round-to-nearest-even
  return (unsigned short)(u >> 16);
}
__device__ __forceinline__ void atomAddF(float* p, float v) { unsafeAtomicAdd(p, v); }
__device__ __forceinline__ float leaky(float v) { return v > 0.f ? v : 0.2f * v; }

// ---------------------------------------------------------------------------
// H = X @ W with the full 256x256 W resident in LDS (bf16, transposed, padded).
// Each block stages W^T once, then sweeps 64-row X tiles (grid-stride).
// 8 waves: wr = wave&3 -> 16-row tile, wc = wave>>2 -> 128-col half (8 WMMA tiles).
// LDS: wt 256*264*2 = 132KB, xs 64*264*2 = 33KB  (CDNA5: up to 320KB/WG).
// ---------------------------------------------------------------------------
__launch_bounds__(256)
__global__ void gcn_gemm_wmma(const float* __restrict__ X, const float* __restrict__ W,
                              float* __restrict__ H, int nrows, int ntiles) {
  __shared__ unsigned short wt[DIM * WT_STR];  // wt[n*WT_STR + k] = W[k][n]
  __shared__ unsigned short xs[64 * XS_STR];   // xs[r*XS_STR + k]

  const int tid = threadIdx.x;

  // Stage full W transposed -> bf16 LDS (coalesced global reads)
  for (int idx = tid; idx < DIM * DIM; idx += 256) {
    int k = idx >> 8, n = idx & 255;
    wt[n * WT_STR + k] = f32_to_bf16(W[idx]);
  }

  const int wave  = tid >> 5;
  const int lane  = tid & 31;
  const int lhalf = lane >> 4;        // 0: lanes 0-15, 1: lanes 16-31
  const int l16   = lane & 15;
  const int wr    = wave & 3;         // row tile (16 rows) within 64-row block
  const int wc    = wave >> 2;        // col half: columns [wc*128, wc*128+128)

  for (int t = blockIdx.x; t < ntiles; t += gridDim.x) {
    const int row0 = t * 64;
    __syncthreads();   // wt ready (1st iter) / previous tile's readers done
    for (int idx = tid; idx < 64 * DIM; idx += 256) {
      int r = idx >> 8, c = idx & 255;
      int gr = row0 + r;
      float v = (gr < nrows) ? X[(size_t)gr * DIM + c] : 0.0f;
      xs[r * XS_STR + c] = f32_to_bf16(v);
    }
    __syncthreads();

    v8f acc[8] = {};
    const unsigned short* xr = xs + (wr * 16 + l16) * XS_STR;

    for (int k0 = 0; k0 < DIM; k0 += 32) {
      // A fragment: lo-half lanes K {0..15}, hi-half K {16..31}
      BF16Frag af;
      af.lo = *(const ush8*)(xr + k0 + 16 * lhalf);
      af.hi = *(const ush8*)(xr + k0 + 16 * lhalf + 8);
      v16bf a = __builtin_bit_cast(v16bf, af);

#pragma unroll
      for (int c = 0; c < 8; ++c) {
        // B fragment: element i <-> K = k0 + 16*lhalf + i, contiguous in W^T row
        const unsigned short* wp = wt + (wc * 128 + c * 16 + l16) * WT_STR + k0 + 16 * lhalf;
        BF16Frag bf;
        bf.lo = *(const ush8*)(wp);
        bf.hi = *(const ush8*)(wp + 8);
        v16bf b = __builtin_bit_cast(v16bf, bf);
        acc[c] = __builtin_amdgcn_wmma_f32_16x16x32_bf16(false, a, false, b, (short)0, acc[c], false, false);
      }
    }

    // C/D layout: VGPR r -> M = r + 8*lhalf, N = l16
    const int growbase = row0 + wr * 16 + 8 * lhalf;
    const int gcol     = wc * 128 + l16;
#pragma unroll
    for (int r = 0; r < 8; ++r) {
      int grow = growbase + r;
      if (grow < nrows) {
        float* hp = H + (size_t)grow * DIM + gcol;
#pragma unroll
        for (int c = 0; c < 8; ++c) hp[c * 16] = acc[c][r];
      }
    }
  }
}

// ---------------------------------------------------------------------------
__global__ void zero_f32(float* __restrict__ p, long n) {
  long i = (long)blockIdx.x * blockDim.x + threadIdx.x;
  long stride = (long)gridDim.x * blockDim.x;
  for (; i < n; i += stride) p[i] = 0.0f;
}

__global__ void init_deg(float* __restrict__ deg, int n) {
  int i = blockIdx.x * blockDim.x + threadIdx.x;
  if (i < n) deg[i] = 1.0f;   // self-loop
}

__global__ void deg_edges(const int* __restrict__ dst, float* __restrict__ deg, int ne) {
  int i = blockIdx.x * blockDim.x + threadIdx.x;
  int stride = gridDim.x * blockDim.x;
  for (; i < ne; i += stride) atomAddF(&deg[dst[i]], 1.0f);
}

__global__ void deg_to_rsqrt(float* __restrict__ deg, int n) {
  int i = blockIdx.x * blockDim.x + threadIdx.x;
  if (i < n) deg[i] = rsqrtf(deg[i]);
}

// One wave32 per edge: gather h[src] (L2-resident, 102MB < 192MB L2),
// scale by inv_sqrt[src], scatter-add into agg[dst] with hardware f32 atomics.
__global__ void gcn_scatter(const float* __restrict__ h, const float* __restrict__ inv,
                            const int* __restrict__ src, const int* __restrict__ dst,
                            float* __restrict__ agg, int ne) {
  int gw   = (blockIdx.x * blockDim.x + threadIdx.x) >> 5;
  int lane = threadIdx.x & 31;
  int nw   = (gridDim.x * blockDim.x) >> 5;
  for (int e = gw; e < ne; e += nw) {
    int s = src[e], d = dst[e];
    float w = inv[s];
    const float4* hs = (const float4*)(h + (size_t)s * DIM);
    float* ad = agg + (size_t)d * DIM;
#pragma unroll
    for (int c = 0; c < 2; ++c) {
      float4 v = hs[lane + 32 * c];
      int b = (lane + 32 * c) * 4;
      atomAddF(ad + b + 0, v.x * w);
      atomAddF(ad + b + 1, v.y * w);
      atomAddF(ad + b + 2, v.z * w);
      atomAddF(ad + b + 3, v.w * w);
    }
  }
}

// x_next = leaky( agg*inv + h*inv^2 + bias ), in-place over h, float4-vectorized.
__global__ void gcn_combine(float4* __restrict__ h, const float4* __restrict__ agg,
                            const float* __restrict__ inv, const float4* __restrict__ bias4) {
  size_t idx = (size_t)blockIdx.x * blockDim.x + threadIdx.x;   // over N * 64
  int i = (int)(idx >> 6);
  int q = (int)(idx & 63);
  float iv = inv[i];
  float s  = iv * iv;
  float4 a = agg[idx], x = h[idx], b = bias4[q];
  float4 o;
  o.x = leaky(a.x * iv + x.x * s + b.x);
  o.y = leaky(a.y * iv + x.y * s + b.y);
  o.z = leaky(a.z * iv + x.z * s + b.z);
  o.w = leaky(a.w * iv + x.w * s + b.w);
  h[idx] = o;
}

__global__ void count_nodes(const int* __restrict__ batch, float* __restrict__ counts, int n) {
  int i = blockIdx.x * blockDim.x + threadIdx.x;
  if (i < n) atomAddF(&counts[batch[i]], 1.0f);
}

// batch is sorted: run-length partial sums, one atomic per (block, graph-run, dim).
__global__ void pool_sum(const float* __restrict__ x, const int* __restrict__ batch,
                         float* __restrict__ pooled, int n) {
  int base = blockIdx.x * NPB;
  if (base >= n) return;
  int d   = threadIdx.x;                 // 0..255
  int end = base + NPB; if (end > n) end = n;
  float s = 0.f;
  int g = batch[base];
  for (int i = base; i < end; ++i) {
    int gi = batch[i];
    if (gi != g) { atomAddF(&pooled[g * DIM + d], s); s = 0.f; g = gi; }
    s += x[(size_t)i * DIM + d];
  }
  atomAddF(&pooled[g * DIM + d], s);
}

__global__ void mlp1(const float* __restrict__ pooled, const float* __restrict__ counts,
                     const float* __restrict__ w, const float* __restrict__ b,
                     float* __restrict__ h1) {
  int g = blockIdx.x;        // 0..63
  int j = threadIdx.x;       // 0..127
  float invc = 1.0f / fmaxf(counts[g], 1.0f);
  float acc = b[j];
  for (int k = 0; k < DIM; ++k)
    acc += pooled[g * DIM + k] * invc * w[k * 128 + j];
  h1[g * 128 + j] = leaky(acc);
}

__global__ void mlp2(const float* __restrict__ h1, const float* __restrict__ w,
                     const float* __restrict__ b, float* __restrict__ out) {
  int t = blockIdx.x * blockDim.x + threadIdx.x;   // 128 = 64 graphs * 2 classes
  if (t >= NGRAPH * 2) return;
  int g = t >> 1, c = t & 1;
  float acc = b[c];
  for (int k = 0; k < 128; ++k) acc += h1[g * 128 + k] * w[k * 2 + c];
  out[g * 2 + c] = acc;
}

// ---------------------------------------------------------------------------
extern "C" void kernel_launch(void* const* d_in, const int* in_sizes, int n_in,
                              void* d_out, int out_size, void* d_ws, size_t ws_size,
                              hipStream_t stream) {
  const float* x      = (const float*)d_in[0];
  const float* conv_w = (const float*)d_in[1];
  const float* conv_b = (const float*)d_in[2];
  const float* lin1_w = (const float*)d_in[3];
  const float* lin1_b = (const float*)d_in[4];
  const float* lin2_w = (const float*)d_in[5];
  const float* lin2_b = (const float*)d_in[6];
  const int*   eidx   = (const int*)d_in[7];
  const int*   batch  = (const int*)d_in[8];

  const int N = in_sizes[8];                 // 100000
  const int E = in_sizes[7] / 2;             // 1600000
  const int L = in_sizes[1] / (DIM * DIM);   // 3
  const int* src = eidx;
  const int* dst = eidx + E;

  const size_t feat = (size_t)N * DIM;
  char* w = (char*)d_ws;
  float* buf0   = (float*)w;  w += feat * sizeof(float);
  float* buf1   = (float*)w;  w += feat * sizeof(float);
  float* agg    = (float*)w;  w += feat * sizeof(float);
  float* invsq  = (float*)w;  w += (size_t)N * sizeof(float);
  float* pooled = (float*)w;  w += (size_t)NGRAPH * DIM * sizeof(float);
  float* counts = (float*)w;  w += (size_t)NGRAPH * sizeof(float);
  float* h1     = (float*)w;  w += (size_t)NGRAPH * 128 * sizeof(float);
  (void)ws_size; (void)n_in;

  const int nblk   = (N + 255) / 256;
  const int ntiles = (N + 63) / 64;

  // Degree + symmetric normalization
  init_deg<<<nblk, 256, 0, stream>>>(invsq, N);
  deg_edges<<<2048, 256, 0, stream>>>(dst, invsq, E);
  deg_to_rsqrt<<<nblk, 256, 0, stream>>>(invsq, N);

  // 3 GCN layers (ping-pong buf0/buf1; combine is in-place on GEMM output)
  for (int l = 0; l < L; ++l) {
    const float* in = (l == 0) ? x : ((l == 1) ? buf0 : buf1);
    float* h        = (l == 1) ? buf1 : buf0;
    gcn_gemm_wmma<<<256, 256, 0, stream>>>(in, conv_w + (size_t)l * DIM * DIM, h, N, ntiles);
    zero_f32<<<4096, 256, 0, stream>>>(agg, (long)feat);
    gcn_scatter<<<4096, 256, 0, stream>>>(h, invsq, src, dst, agg, E);
    gcn_combine<<<(N * (DIM / 4)) / 256, 256, 0, stream>>>(
        (float4*)h, (const float4*)agg, invsq, (const float4*)(conv_b + (size_t)l * DIM));
  }
  float* feats = buf0;   // after L=3 layers result is in buf0

  // Global mean pool + MLP head
  zero_f32<<<64, 256, 0, stream>>>(pooled, (long)NGRAPH * DIM);
  zero_f32<<<1, 64, 0, stream>>>(counts, (long)NGRAPH);
  count_nodes<<<nblk, 256, 0, stream>>>(batch, counts, N);
  pool_sum<<<(N + NPB - 1) / NPB, 256, 0, stream>>>(feats, batch, pooled, N);
  mlp1<<<NGRAPH, 128, 0, stream>>>(pooled, counts, lin1_w, lin1_b, h1);
  mlp2<<<1, 128, 0, stream>>>(h1, lin2_w, lin2_b, (float*)d_out);
}
